// TopKSelectorOTT_2662879723640
// MI455X (gfx1250) — matrix operand order repaired and checked by reading, not compiled, not run on hardware
//
#include <hip/hip_runtime.h>

// ---------- types ----------
typedef __attribute__((ext_vector_type(16))) __bf16 v16bf;
typedef __attribute__((ext_vector_type(8)))  __bf16 bf16x8;
typedef __attribute__((ext_vector_type(8)))  float  v8f;

#define SK_EPS   0.01f
#define SK_IEPS  100.0f
#define SK_LOGN  4.1588830833596715f   // log(64)

__device__ __forceinline__ unsigned short f2bf(float x) {
  union { float f; unsigned int u; } v; v.f = x;
  unsigned int r = v.u + 0x7FFFu + ((v.u >> 16) & 1u);
  return (unsigned short)(r >> 16);
}

// ---------- conversion kernels ----------
__global__ void cvt_bf16(const float* __restrict__ src, unsigned short* __restrict__ dst, int n) {
  int i = blockIdx.x * blockDim.x + threadIdx.x;
  if (i < n) dst[i] = f2bf(src[i]);
}

// dst[c*R + r] = bf16(src[r*C + c])   (store transposed)
__global__ void cvt_bf16_t(const float* __restrict__ src, unsigned short* __restrict__ dst, int R, int C) {
  int i = blockIdx.x * blockDim.x + threadIdx.x;
  if (i < R * C) {
    int r = i / C, c = i - r * C;
    dst[c * R + r] = f2bf(src[i]);
  }
}

// ---------- WMMA helpers ----------
__device__ __forceinline__ v16bf cat16(bf16x8 lo, bf16x8 hi) {
  return __builtin_shufflevector(lo, hi, 0,1,2,3,4,5,6,7,8,9,10,11,12,13,14,15);
}
// A fragment 16x32 bf16: base = row ptr (+8 elems if lane>=16); halves at +0 and +16.
__device__ __forceinline__ v16bf load_a(const unsigned short* p) {
  bf16x8 lo = *(const bf16x8*)(p);
  bf16x8 hi = *(const bf16x8*)(p + 16);
  return cat16(lo, hi);
}
// B fragment 32x16 bf16 from transposed weights: base = col ptr (+8 elems if lane>=16); halves at +0 and +16.
__device__ __forceinline__ v16bf load_b(const unsigned short* p) {
  bf16x8 lo = *(const bf16x8*)(p);
  bf16x8 hi = *(const bf16x8*)(p + 16);
  return cat16(lo, hi);
}

// ---------- GEMM1: h = relu(x @ W1 + b1), bf16 in, bf16 out ----------
// xb [16384][512] bf16, w1t [1024][512] bf16 (W1 transposed), hb [16384][1024] bf16
__global__ __launch_bounds__(256) void gemm1_relu(const unsigned short* __restrict__ xb,
                                                  const unsigned short* __restrict__ w1t,
                                                  const float* __restrict__ b1,
                                                  unsigned short* __restrict__ hb) {
  const int lane = threadIdx.x & 31;
  const int wave = threadIdx.x >> 5;
  const int wm = wave & 3, wn = wave >> 2;
  const int m0 = blockIdx.x * 128 + wm * 32;
  const int n0 = blockIdx.y * 64 + wn * 32;
  const int r  = lane & 15;
  const int hi = (lane >= 16);
  const int KD = 512;

  v8f c00 = {0,0,0,0,0,0,0,0}, c01 = c00, c10 = c00, c11 = c00;

  const unsigned short* a0 = xb  + (size_t)(m0 +      r) * KD + (hi ? 8 : 0);
  const unsigned short* a1 = xb  + (size_t)(m0 + 16 + r) * KD + (hi ? 8 : 0);
  const unsigned short* p0 = w1t + (size_t)(n0 +      r) * KD + (hi ? 8 : 0);
  const unsigned short* p1 = w1t + (size_t)(n0 + 16 + r) * KD + (hi ? 8 : 0);

  for (int k = 0; k < KD; k += 32) {
    v16bf A0 = load_a(a0 + k);
    v16bf A1 = load_a(a1 + k);
    v16bf B0 = load_b(p0 + k);
    v16bf B1 = load_b(p1 + k);
    c00 = __builtin_amdgcn_wmma_f32_16x16x32_bf16(false, A0, false, B0, (short)0, c00, false, false);
    c01 = __builtin_amdgcn_wmma_f32_16x16x32_bf16(false, A0, false, B1, (short)0, c01, false, false);
    c10 = __builtin_amdgcn_wmma_f32_16x16x32_bf16(false, A1, false, B0, (short)0, c10, false, false);
    c11 = __builtin_amdgcn_wmma_f32_16x16x32_bf16(false, A1, false, B1, (short)0, c11, false, false);
  }

  // C/D layout: lanes 0-15: N=lane, M=vgpr; lanes 16-31: N=lane-16, M=8+vgpr
  const int mofs = hi ? 8 : 0;
  {
    int n = n0 + r; float bias = b1[n];
    #pragma unroll
    for (int q = 0; q < 8; q++) {
      hb[(size_t)(m0 +      q + mofs) * 1024 + n] = f2bf(fmaxf(c00[q] + bias, 0.f));
      hb[(size_t)(m0 + 16 + q + mofs) * 1024 + n] = f2bf(fmaxf(c10[q] + bias, 0.f));
    }
  }
  {
    int n = n0 + 16 + r; float bias = b1[n];
    #pragma unroll
    for (int q = 0; q < 8; q++) {
      hb[(size_t)(m0 +      q + mofs) * 1024 + n] = f2bf(fmaxf(c01[q] + bias, 0.f));
      hb[(size_t)(m0 + 16 + q + mofs) * 1024 + n] = f2bf(fmaxf(c11[q] + bias, 0.f));
    }
  }
}

// ---------- GEMM2: scores = h @ W2 + b2, bf16 in, f32 out ----------
// hb [16384][1024] bf16, w2t [64][1024] bf16, out [16384][64] f32
__global__ __launch_bounds__(256) void gemm2(const unsigned short* __restrict__ hb,
                                             const unsigned short* __restrict__ w2t,
                                             const float* __restrict__ b2,
                                             float* __restrict__ out) {
  const int lane = threadIdx.x & 31;
  const int wave = threadIdx.x >> 5;
  const int wm = wave & 3, wn = wave >> 2;
  const int m0 = blockIdx.x * 128 + wm * 32;
  const int n0 = wn * 32;
  const int r  = lane & 15;
  const int hi = (lane >= 16);
  const int KD = 1024;

  v8f c00 = {0,0,0,0,0,0,0,0}, c01 = c00, c10 = c00, c11 = c00;

  const unsigned short* a0 = hb  + (size_t)(m0 +      r) * KD + (hi ? 8 : 0);
  const unsigned short* a1 = hb  + (size_t)(m0 + 16 + r) * KD + (hi ? 8 : 0);
  const unsigned short* p0 = w2t + (size_t)(n0 +      r) * KD + (hi ? 8 : 0);
  const unsigned short* p1 = w2t + (size_t)(n0 + 16 + r) * KD + (hi ? 8 : 0);

  for (int k = 0; k < KD; k += 32) {
    v16bf A0 = load_a(a0 + k);
    v16bf A1 = load_a(a1 + k);
    v16bf B0 = load_b(p0 + k);
    v16bf B1 = load_b(p1 + k);
    c00 = __builtin_amdgcn_wmma_f32_16x16x32_bf16(false, A0, false, B0, (short)0, c00, false, false);
    c01 = __builtin_amdgcn_wmma_f32_16x16x32_bf16(false, A0, false, B1, (short)0, c01, false, false);
    c10 = __builtin_amdgcn_wmma_f32_16x16x32_bf16(false, A1, false, B0, (short)0, c10, false, false);
    c11 = __builtin_amdgcn_wmma_f32_16x16x32_bf16(false, A1, false, B1, (short)0, c11, false, false);
  }

  const int mofs = hi ? 8 : 0;
  {
    int n = n0 + r; float bias = b2[n];
    #pragma unroll
    for (int q = 0; q < 8; q++) {
      out[(size_t)(m0 +      q + mofs) * 64 + n] = c00[q] + bias;
      out[(size_t)(m0 + 16 + q + mofs) * 64 + n] = c10[q] + bias;
    }
  }
  {
    int n = n0 + 16 + r; float bias = b2[n];
    #pragma unroll
    for (int q = 0; q < 8; q++) {
      out[(size_t)(m0 +      q + mofs) * 64 + n] = c01[q] + bias;
      out[(size_t)(m0 + 16 + q + mofs) * 64 + n] = c11[q] + bias;
    }
  }
}

// ---------- Sinkhorn soft-rank + top-k soft mask (in place on scores) ----------
template <bool IS_MAX>
__device__ __forceinline__ float blockRed64(float v, float* red2) {
  #pragma unroll
  for (int o = 16; o > 0; o >>= 1) {
    float ov = __shfl_xor(v, o, 32);
    v = IS_MAX ? fmaxf(v, ov) : (v + ov);
  }
  __syncthreads();                       // protect red2 / LDS reuse from previous phase
  if ((threadIdx.x & 31) == 0) red2[threadIdx.x >> 5] = v;
  __syncthreads();
  return IS_MAX ? fmaxf(red2[0], red2[1]) : (red2[0] + red2[1]);
}

__global__ __launch_bounds__(64) void sinkhorn_mask(float* __restrict__ io) {
  __shared__ float Kmat[64 * 65];   // padded stride 65 -> conflict-free rows AND columns
  __shared__ float ex[64];
  __shared__ float garr[64];
  __shared__ float red2[2];

  const int t = threadIdx.x;
  const size_t base = (size_t)blockIdx.x * 64;
  float s = io[base + t];

  // mean / std (ddof=0) / sigmoid squash
  float mu  = blockRed64<false>(s, red2) * (1.f / 64.f);
  float d   = s - mu;
  float var = blockRed64<false>(d * d, red2) * (1.f / 64.f);
  float sd  = sqrtf(var) + 1e-8f;
  float z   = 1.f / (1.f + __expf(-(s - mu) / sd));

  // K[i][j] = exp(-(z_i - y_j)^2 / eps), row per thread
  {
    float* Kr = &Kmat[t * 65];
    #pragma unroll 8
    for (int j = 0; j < 64; j++) {
      float dy = z - (float)j * (1.f / 63.f);
      Kr[j] = __expf(-SK_IEPS * dy * dy);
    }
  }
  __syncthreads();

  // 50 Sinkhorn iterations (f-update then g-update), factored log-domain:
  // LSE_j((g_j - C_ij)/eps) = gmax/eps + log( sum_j K_ij * exp((g_j-gmax)/eps) )
  float f = 0.f, g = 0.f;
  for (int it = 0; it < 50; ++it) {
    float gmax = blockRed64<true>(g, red2);
    ex[t] = __expf((g - gmax) * SK_IEPS);
    __syncthreads();
    float S = 0.f;
    const float* Kr = &Kmat[t * 65];
    #pragma unroll 8
    for (int j = 0; j < 64; j++) S = fmaf(Kr[j], ex[j], S);
    f = -SK_EPS * SK_LOGN - gmax - SK_EPS * __logf(fmaxf(S, 1e-30f));

    float fmx = blockRed64<true>(f, red2);
    ex[t] = __expf((f - fmx) * SK_IEPS);
    __syncthreads();
    float T = 0.f;
    #pragma unroll 8
    for (int i = 0; i < 64; i++) T = fmaf(Kmat[i * 65 + t], ex[i], T);
    g = -SK_EPS * SK_LOGN - fmx - SK_EPS * __logf(fmaxf(T, 1e-30f));
  }

  garr[t] = g;
  __syncthreads();

  // ranks_i = n * sum_j exp((f_i + g_j - C_ij)/eps) * j  (per-row max-stabilized)
  float m = -3.4e38f;
  #pragma unroll 8
  for (int j = 0; j < 64; j++) {
    float dy = z - (float)j * (1.f / 63.f);
    m = fmaxf(m, f + garr[j] - dy * dy);
  }
  float sum = 0.f;
  #pragma unroll 8
  for (int j = 0; j < 64; j++) {
    float dy = z - (float)j * (1.f / 63.f);
    sum = fmaf(__expf((f + garr[j] - dy * dy - m) * SK_IEPS), (float)j, sum);
  }
  float rank = 64.f * __expf(m * SK_IEPS) * sum;

  float inv  = 63.f - rank;                       // n-1 - rank
  float mask = 1.f / (1.f + __expf(-(inv - 56.f) * 10.f));  // threshold n-K=56, steepness 10
  io[base + t] = mask;
}

// ---------- launcher ----------
extern "C" void kernel_launch(void* const* d_in, const int* in_sizes, int n_in,
                              void* d_out, int out_size, void* d_ws, size_t ws_size,
                              hipStream_t stream) {
  const float* x  = (const float*)d_in[0];   // [16384,512]
  const float* W1 = (const float*)d_in[1];   // [512,1024]
  const float* b1 = (const float*)d_in[2];   // [1024]
  const float* W2 = (const float*)d_in[3];   // [1024,64]
  const float* b2 = (const float*)d_in[4];   // [64]
  float* out = (float*)d_out;                // [16384,64]

  char* ws = (char*)d_ws;
  unsigned short* xb  = (unsigned short*)(ws);              // 16384*512*2  = 16,777,216 B
  unsigned short* w1t = (unsigned short*)(ws + 16777216);   // 1024*512*2   =  1,048,576 B
  unsigned short* w2t = (unsigned short*)(ws + 17825792);   // 64*1024*2    =    131,072 B
  unsigned short* hb  = (unsigned short*)(ws + 17956864);   // 16384*1024*2 = 33,554,432 B
  // total ws usage: 51,511,296 B

  cvt_bf16  <<<(8388608 + 255) / 256, 256, 0, stream>>>(x,  xb, 8388608);
  cvt_bf16_t<<<(524288  + 255) / 256, 256, 0, stream>>>(W1, w1t, 512, 1024);
  cvt_bf16_t<<<(65536   + 255) / 256, 256, 0, stream>>>(W2, w2t, 1024, 64);

  gemm1_relu<<<dim3(128, 16), 256, 0, stream>>>(xb, w1t, b1, hb);
  gemm2     <<<dim3(128, 1),  256, 0, stream>>>(hb, w2t, b2, out);

  sinkhorn_mask<<<16384, 64, 0, stream>>>(out);
}
